// EncoderBlock_84937273246256
// MI455X (gfx1250) — compile-verified, matrix operand order checked by reading, not run on hardware
//
#include <hip/hip_runtime.h>
#include <hip/hip_bf16.h>

typedef __bf16 bf16_t;
typedef __attribute__((ext_vector_type(16))) __bf16 bf16x16;
typedef __attribute__((ext_vector_type(8)))  float  f32x8;

#define LN_EPS 1e-5f

enum { MODE_F32 = 0, MODE_BF16 = 1, MODE_BF16_T = 2, MODE_BIAS_RELU_BF16 = 3, MODE_BIAS_F32 = 4 };

// ---------------------------------------------------------------------------
// CDNA5 async global->LDS DMA (no VGPR staging, tracked by ASYNCcnt)
// ---------------------------------------------------------------------------
__device__ __forceinline__ void async_ld_b128(unsigned lds_addr, const void* gaddr) {
    asm volatile("global_load_async_to_lds_b128 %0, %1, off"
                 :: "v"(lds_addr), "v"(gaddr) : "memory");
}
__device__ __forceinline__ void wait_asynccnt0() {
    asm volatile("s_wait_asynccnt 0" ::: "memory");
}
__device__ __forceinline__ unsigned lds_off(const void* p) {
    return (unsigned)(uintptr_t)p;   // LDS aperture: low 32 bits = LDS byte address
}

// ---------------------------------------------------------------------------
// Elementwise f32 -> bf16 convert
// ---------------------------------------------------------------------------
__global__ void cvt_f32_bf16(const float* __restrict__ in, bf16_t* __restrict__ out, long long n) {
    long long i = (long long)blockIdx.x * blockDim.x + threadIdx.x;
    long long stride = (long long)gridDim.x * blockDim.x;
    for (; i < n; i += stride) out[i] = (bf16_t)in[i];
}

// ---------------------------------------------------------------------------
// Transpose + convert: in R x C f32 row-major -> out C x R bf16 row-major
// ---------------------------------------------------------------------------
__global__ __launch_bounds__(256) void transpose_cvt(const float* __restrict__ in,
                                                     bf16_t* __restrict__ out,
                                                     int R, int C) {
    __shared__ float t[32][33];
    const int cb = blockIdx.x * 32, rb = blockIdx.y * 32;
    const int tx = threadIdx.x, ty = threadIdx.y;   // 32 x 8
#pragma unroll
    for (int k = 0; k < 4; ++k)
        t[ty + 8 * k][tx] = in[(long long)(rb + ty + 8 * k) * C + cb + tx];
    __syncthreads();
#pragma unroll
    for (int k = 0; k < 4; ++k)
        out[(long long)(cb + ty + 8 * k) * R + rb + tx] = (bf16_t)t[tx][ty + 8 * k];
}

// ---------------------------------------------------------------------------
// bf16 WMMA GEMM:  C[M,N] = A[M,K] (row-major) * Bt[N,K]^T  (Bt row-major N x K)
// 128x128 workgroup tile, K-step 32, double-buffered LDS fed by
// global_load_async_to_lds_b128; 8 waves, each computes 32x64 (2x4 WMMA accums).
// ---------------------------------------------------------------------------
#define TM 128
#define TN 128
#define TK 32
#define LDT 40   // padded LDS row stride (bf16 elems): 80B -> conflict-free frag reads

template <int MODE>
__global__ __launch_bounds__(256) void gemm_bf16_wmma(
    const bf16_t* __restrict__ A, const bf16_t* __restrict__ Bt,
    float* __restrict__ Cf, bf16_t* __restrict__ Cb,
    const float* __restrict__ bias,
    int M, int N, int K,
    long long sA, long long sB, long long sC)
{
    __shared__ __align__(16) bf16_t As[2][TM * LDT];
    __shared__ __align__(16) bf16_t Bs[2][TM * LDT];

    const int tid  = threadIdx.x;
    const int lane = tid & 31;
    const int wave = tid >> 5;
    const int wm = wave >> 1;          // 0..3
    const int wn = wave & 1;           // 0..1
    const int mb = wm * 32, nb = wn * 64;

    const int z = blockIdx.z;
    const long long m0 = (long long)blockIdx.y * TM;
    const long long n0 = (long long)blockIdx.x * TN;

    const bf16_t* Ab = A + (long long)z * sA;
    const bf16_t* Bb = Bt + (long long)z * sB;

    // tile staging: 128 rows x 32 bf16 = 512 x 16B; each thread DMAs 2 x b128 per operand
    const int lrow = tid >> 1;             // 0..127
    const int lce  = (tid & 1) * 16;       // element col: 0 or 16 (covers 2 x 8 bf16)

    const bf16_t* gA = Ab + (m0 + lrow) * (long long)K + lce;
    const bf16_t* gB = Bb + (n0 + lrow) * (long long)K + lce;
    const unsigned ldsA = lds_off(&As[0][lrow * LDT + lce]);   // buffer-0 dest
    const unsigned ldsB = lds_off(&Bs[0][lrow * LDT + lce]);
    const unsigned bufStep = TM * LDT * (unsigned)sizeof(bf16_t);   // bytes buf0 -> buf1

    const int KT = K / TK;

    // prologue: DMA tile 0 into buffer 0
    async_ld_b128(ldsA,      gA);
    async_ld_b128(ldsA + 16, gA + 8);
    async_ld_b128(ldsB,      gB);
    async_ld_b128(ldsB + 16, gB + 8);

    // loop-carried prefetch pointers (next tile = kt+1)
    const bf16_t* pA = gA + TK;
    const bf16_t* pB = gB + TK;

    f32x8 acc[2][4];
#pragma unroll
    for (int i = 0; i < 2; ++i)
#pragma unroll
        for (int j = 0; j < 4; ++j)
#pragma unroll
            for (int e = 0; e < 8; ++e) acc[i][j][e] = 0.f;

    const int fr = lane & 15;          // row within 16x16 fragment
    const int fk = (lane >> 4) * 16;   // K-half element offset

    for (int kt = 0; kt < KT; ++kt) {
        wait_asynccnt0();      // this wave's DMA for tile kt has landed
        __syncthreads();       // all waves' tiles landed; all done reading other buf

        const int buf = kt & 1;
        if (kt + 1 < KT) {     // DMA tile kt+1 into the other buffer
            const unsigned dst = ((kt + 1) & 1) ? bufStep : 0u;
            async_ld_b128(ldsA + dst,      pA);
            async_ld_b128(ldsA + dst + 16, pA + 8);
            async_ld_b128(ldsB + dst,      pB);
            async_ld_b128(ldsB + dst + 16, pB + 8);
            pA += TK;
            pB += TK;
        }

        union FB { bf16x16 v; uint4 q[2]; };
        FB a[2], b[4];
#pragma unroll
        for (int i = 0; i < 2; ++i) {
            const uint4* p = (const uint4*)&As[buf][(mb + i * 16 + fr) * LDT + fk];
            a[i].q[0] = p[0]; a[i].q[1] = p[1];
        }
#pragma unroll
        for (int j = 0; j < 4; ++j) {
            const uint4* p = (const uint4*)&Bs[buf][(nb + j * 16 + fr) * LDT + fk];
            b[j].q[0] = p[0]; b[j].q[1] = p[1];
        }

#pragma unroll
        for (int i = 0; i < 2; ++i)
#pragma unroll
            for (int j = 0; j < 4; ++j)
                acc[i][j] = __builtin_amdgcn_wmma_f32_16x16x32_bf16(
                    false, a[i].v, false, b[j].v, (short)0, acc[i][j], false, false);
    }

    // ---- epilogue (MODE is compile-time) ----
    const long long cb0 = (long long)z * sC;
#pragma unroll
    for (int i = 0; i < 2; ++i) {
#pragma unroll
        for (int j = 0; j < 4; ++j) {
            const long long col  = n0 + nb + j * 16 + (lane & 15);
            const long long rowb = m0 + mb + i * 16 + (lane >> 4) * 8;
            float bv = 0.f;
            if (MODE == MODE_BIAS_RELU_BF16 || MODE == MODE_BIAS_F32) bv = bias[col];
#pragma unroll
            for (int e = 0; e < 8; ++e) {
                float v = acc[i][j][e] + bv;
                const long long row = rowb + e;
                if (MODE == MODE_BIAS_RELU_BF16) v = v > 0.f ? v : 0.f;
                if (MODE == MODE_F32 || MODE == MODE_BIAS_F32)
                    Cf[cb0 + row * (long long)N + col] = v;
                else if (MODE == MODE_BF16_T)
                    Cb[cb0 + col * (long long)M + row] = (bf16_t)v;
                else
                    Cb[cb0 + row * (long long)N + col] = (bf16_t)v;
            }
        }
    }
}

// ---------------------------------------------------------------------------
// Row softmax over 2048 cols (scale folded in), f32 in -> bf16 out.
// 256 threads -> exactly 8 elements/thread kept in registers: one HBM read,
// one exp per element.
// ---------------------------------------------------------------------------
__global__ __launch_bounds__(256) void softmax_rows(const float* __restrict__ S,
                                                    bf16_t* __restrict__ O,
                                                    float scale) {
    __shared__ float red[256];
    const int tid = threadIdx.x;
    const long long base = (long long)blockIdx.x * 2048;

    float v[8];
    float lmax = -3.4e38f;
#pragma unroll
    for (int r = 0; r < 8; ++r) {
        v[r] = S[base + tid + r * 256] * scale;
        lmax = fmaxf(lmax, v[r]);
    }
    red[tid] = lmax; __syncthreads();
    for (int st = 128; st > 0; st >>= 1) {
        if (tid < st) red[tid] = fmaxf(red[tid], red[tid + st]);
        __syncthreads();
    }
    const float m = red[0];
    __syncthreads();

    float lsum = 0.f;
#pragma unroll
    for (int r = 0; r < 8; ++r) {
        v[r] = __expf(v[r] - m);
        lsum += v[r];
    }
    red[tid] = lsum; __syncthreads();
    for (int st = 128; st > 0; st >>= 1) {
        if (tid < st) red[tid] += red[tid + st];
        __syncthreads();
    }
    const float inv = 1.f / red[0];
#pragma unroll
    for (int r = 0; r < 8; ++r)
        O[base + tid + r * 256] = (bf16_t)(v[r] * inv);
}

// ---------------------------------------------------------------------------
// out = LayerNorm(x + y) * g + b over D=1024; optional bf16 copy.
// 256 threads -> exactly 4 elements/thread in registers.
// ---------------------------------------------------------------------------
__global__ __launch_bounds__(256) void add_ln(const float* __restrict__ x,
                                              const float* __restrict__ y,
                                              const float* __restrict__ g,
                                              const float* __restrict__ b,
                                              float* __restrict__ outF,
                                              bf16_t* __restrict__ outB) {
    __shared__ float r1[256];
    __shared__ float r2[256];
    const int tid = threadIdx.x;
    const long long base = (long long)blockIdx.x * 1024;

    float v[4];
    float s = 0.f, s2 = 0.f;
#pragma unroll
    for (int r = 0; r < 4; ++r) {
        const long long i = base + tid + r * 256;
        v[r] = x[i] + y[i];
        s += v[r]; s2 += v[r] * v[r];
    }
    r1[tid] = s; r2[tid] = s2; __syncthreads();
    for (int st = 128; st > 0; st >>= 1) {
        if (tid < st) { r1[tid] += r1[tid + st]; r2[tid] += r2[tid + st]; }
        __syncthreads();
    }
    const float mean = r1[0] * (1.f / 1024.f);
    const float var  = r2[0] * (1.f / 1024.f) - mean * mean;
    const float inv  = rsqrtf(var + LN_EPS);
#pragma unroll
    for (int r = 0; r < 4; ++r) {
        const int c = tid + r * 256;
        const float o = (v[r] - mean) * inv * g[c] + b[c];
        outF[base + c] = o;
        if (outB) outB[base + c] = (bf16_t)o;
    }
}

// ---------------------------------------------------------------------------
// Host launch
// ---------------------------------------------------------------------------
extern "C" void kernel_launch(void* const* d_in, const int* in_sizes, int n_in,
                              void* d_out, int out_size, void* d_ws, size_t ws_size,
                              hipStream_t stream) {
    (void)in_sizes; (void)n_in; (void)out_size; (void)ws_size;

    const int Bn = 4, S = 2048, D = 1024, F = 4096;
    const int M = Bn * S;                 // 8192

    const float* x   = (const float*)d_in[0];
    const float* Wq  = (const float*)d_in[1];
    const float* Wk  = (const float*)d_in[2];
    const float* Wv  = (const float*)d_in[3];
    const float* Wo  = (const float*)d_in[4];
    const float* W1  = (const float*)d_in[5];
    const float* b1  = (const float*)d_in[6];
    const float* W2  = (const float*)d_in[7];
    const float* b2  = (const float*)d_in[8];
    const float* g1  = (const float*)d_in[9];
    const float* be1 = (const float*)d_in[10];
    const float* g2  = (const float*)d_in[11];
    const float* be2 = (const float*)d_in[12];

    // ---- workspace carve-up (regions reused across pipeline phases) ----
    char*  ws  = (char*)d_ws;
    size_t off = 0;
    auto take = [&](size_t bytes) -> void* {
        void* p = ws + off;
        off += (bytes + 255) & ~(size_t)255;
        return p;
    };
    const size_t szW    = (size_t)D * D * 2;       //  2 MB
    const size_t szWff  = (size_t)D * F * 2;       //  8 MB
    const size_t szA16  = (size_t)M * D * 2;       // 16 MB
    const size_t szS32  = (size_t)Bn * S * S * 4;  // 64 MB
    const size_t szS16  = (size_t)Bn * S * S * 2;  // 32 MB

    bf16_t* WQT = (bf16_t*)take(szW);
    bf16_t* WKT = (bf16_t*)take(szW);
    bf16_t* WVT = (bf16_t*)take(szW);
    bf16_t* WOT = (bf16_t*)take(szW);
    bf16_t* W1T = (bf16_t*)take(szWff);
    bf16_t* W2T = (bf16_t*)take(szWff);
    void* R1 = take(szA16);        // xb    -> attb
    void* R2 = take(2 * szA16);    // Qb|Kb -> attproj -> f2
    void* R3 = take(szA16);        // Vt    -> hb
    void* R4 = take(szS32);        // scores-> f1b
    void* R5 = take(szS16);        // wsoft -> h (f32)

    bf16_t* xb      = (bf16_t*)R1;
    bf16_t* attb    = (bf16_t*)R1;
    bf16_t* Qb      = (bf16_t*)R2;
    bf16_t* Kb      = (bf16_t*)((char*)R2 + szA16);
    float*  attproj = (float*)R2;
    float*  f2      = (float*)R2;
    bf16_t* Vt      = (bf16_t*)R3;
    bf16_t* hb      = (bf16_t*)R3;
    float*  scores  = (float*)R4;
    bf16_t* f1b     = (bf16_t*)R4;
    bf16_t* wsoft   = (bf16_t*)R5;
    float*  hbuf    = (float*)R5;

    const dim3 tb(32, 8);

    // ---- precision conversion + weight transposes ----
    cvt_f32_bf16<<<4096, 256, 0, stream>>>(x, xb, (long long)M * D);
    transpose_cvt<<<dim3(D / 32, D / 32), tb, 0, stream>>>(Wq, WQT, D, D);
    transpose_cvt<<<dim3(D / 32, D / 32), tb, 0, stream>>>(Wk, WKT, D, D);
    transpose_cvt<<<dim3(D / 32, D / 32), tb, 0, stream>>>(Wv, WVT, D, D);
    transpose_cvt<<<dim3(D / 32, D / 32), tb, 0, stream>>>(Wo, WOT, D, D);
    transpose_cvt<<<dim3(F / 32, D / 32), tb, 0, stream>>>(W1, W1T, D, F);
    transpose_cvt<<<dim3(D / 32, F / 32), tb, 0, stream>>>(W2, W2T, F, D);

    // ---- Q/K/V projections ----
    gemm_bf16_wmma<MODE_BF16><<<dim3(D / 128, M / 128, 1), 256, 0, stream>>>(
        xb, WQT, nullptr, Qb, nullptr, M, D, D, 0, 0, 0);
    gemm_bf16_wmma<MODE_BF16><<<dim3(D / 128, M / 128, 1), 256, 0, stream>>>(
        xb, WKT, nullptr, Kb, nullptr, M, D, D, 0, 0, 0);
    gemm_bf16_wmma<MODE_BF16_T><<<dim3(D / 128, S / 128, Bn), 256, 0, stream>>>(
        xb, WVT, nullptr, Vt, nullptr, S, D, D,
        (long long)S * D, 0, (long long)D * S);

    // ---- scores = Q @ K^T (per batch) ----
    gemm_bf16_wmma<MODE_F32><<<dim3(S / 128, S / 128, Bn), 256, 0, stream>>>(
        Qb, Kb, scores, nullptr, nullptr, S, S, D,
        (long long)S * D, (long long)S * D, (long long)S * S);

    // ---- softmax(scores / sqrt(dk)) ----
    softmax_rows<<<Bn * S, 256, 0, stream>>>(scores, wsoft, 0.03125f);

    // ---- att = weights @ V (B^T = Vt) ----
    gemm_bf16_wmma<MODE_BF16><<<dim3(D / 128, S / 128, Bn), 256, 0, stream>>>(
        wsoft, Vt, nullptr, attb, nullptr, S, D, S,
        (long long)S * S, (long long)D * S, (long long)S * D);

    // ---- output projection ----
    gemm_bf16_wmma<MODE_F32><<<dim3(D / 128, M / 128, 1), 256, 0, stream>>>(
        attb, WOT, attproj, nullptr, nullptr, M, D, D, 0, 0, 0);

    // ---- h = LN(x + att) ----
    add_ln<<<M, 256, 0, stream>>>(x, attproj, g1, be1, hbuf, hb);

    // ---- FFN ----
    gemm_bf16_wmma<MODE_BIAS_RELU_BF16><<<dim3(F / 128, M / 128, 1), 256, 0, stream>>>(
        hb, W1T, nullptr, f1b, b1, M, F, D, 0, 0, 0);
    gemm_bf16_wmma<MODE_BIAS_F32><<<dim3(D / 128, M / 128, 1), 256, 0, stream>>>(
        f1b, W2T, f2, nullptr, b2, M, D, F, 0, 0, 0);

    // ---- out = LN(h + ffn) ----
    add_ln<<<M, 256, 0, stream>>>(hbuf, f2, g2, be2, (float*)d_out, nullptr);
}
